// AttentionalLSTM_6674379178160
// MI455X (gfx1250) — matrix operand order repaired and checked by reading, not compiled, not run on hardware
//
#include <hip/hip_runtime.h>
#include <stdint.h>
#include <math.h>

// ---- problem constants ----
#define BB   128
#define TT   256
#define DIN  512
#define HDIM 1024
#define KG   1536   // DIN + HDIM (fused gate GEMM K)
#define NG   4096   // 4*HDIM

typedef __attribute__((ext_vector_type(16))) __bf16 v16bf;
typedef __attribute__((ext_vector_type(8)))  float  v8f;

__device__ __forceinline__ unsigned short f2bf(float f) {
  union { float f; unsigned u; } x; x.f = f;
  unsigned u = x.u;
  return (unsigned short)((u + 0x7FFFu + ((u >> 16) & 1u)) >> 16);
}

// ---------------------------------------------------------------------------
// One-time: pack [W_ih | W_hh] -> bf16 Wg[4096][1536], W_attn/W_out -> bf16,
// bias = b_ih + b_hh
// ---------------------------------------------------------------------------
__global__ __launch_bounds__(256) void pack_weights(
    const float* __restrict__ Wih, const float* __restrict__ Whh,
    const float* __restrict__ bih, const float* __restrict__ bhh,
    const float* __restrict__ Wattn, const float* __restrict__ Wout,
    unsigned short* __restrict__ Wg, unsigned short* __restrict__ Wa,
    unsigned short* __restrict__ Wo, float* __restrict__ bias)
{
  const long stride = (long)gridDim.x * 256;
  const long t0 = (long)blockIdx.x * 256 + threadIdx.x;
  for (long i = t0; i < (long)NG * KG; i += stride) {
    int n = (int)(i / KG), k = (int)(i % KG);
    float v = (k < DIN) ? Wih[(long)n * DIN + k] : Whh[(long)n * HDIM + (k - DIN)];
    Wg[i] = f2bf(v);
  }
  for (long i = t0; i < (long)HDIM * HDIM; i += stride) Wa[i] = f2bf(Wattn[i]);
  for (long i = t0; i < (long)HDIM * 2 * HDIM; i += stride) Wo[i] = f2bf(Wout[i]);
  for (long i = t0; i < NG; i += stride) bias[i] = bih[i] + bhh[i];
}

// ---------------------------------------------------------------------------
// One-time: c = c0, xh[:,512:1536] = bf16(h0)
// ---------------------------------------------------------------------------
__global__ __launch_bounds__(256) void init_state(
    const float* __restrict__ h0, const float* __restrict__ c0,
    float* __restrict__ c, unsigned short* __restrict__ xh)
{
  int idx = blockIdx.x * 256 + threadIdx.x;   // 0 .. B*H-1
  int b = idx >> 10, j = idx & 1023;
  c[idx] = c0[idx];
  xh[(long)b * KG + DIN + j] = f2bf(h0[idx]);
}

// ---------------------------------------------------------------------------
// Per step: xh[:,0:512] = bf16(input[:, t, :])
// ---------------------------------------------------------------------------
__global__ __launch_bounds__(256) void prep_x(
    const float* __restrict__ x, int t, unsigned short* __restrict__ xh)
{
  int idx = blockIdx.x * 256 + threadIdx.x;   // 0 .. B*DIN-1
  int b = idx >> 9, j = idx & 511;
  xh[(long)b * KG + j] = f2bf(x[((long)b * TT + t) * DIN + j]);
}

// ---------------------------------------------------------------------------
// bf16 WMMA GEMM: C[M=128][N] = A[128][K] * Bw[N][K]^T  (Bw row-major [N,K])
// Block tile 64x64, 8 waves, each wave 16x32 (2x v_wmma_f32_16x16x32_bf16).
// A tile staged with GLOBAL_LOAD_ASYNC_TO_LDS_B128 (ASYNCcnt path), B tile
// staged transposed via registers; double-buffered LDS, 1 barrier/chunk.
// epi: 0 = +bias -> f32 Cf (gates)   1 = f32 Cf (q)
//      2 = tanh -> d_out[:,t,:], bf16 -> xh h-slot, (last) -> h_f
// ---------------------------------------------------------------------------
#define BM 64
#define BN 64
#define BK 32
#define LDA_S 40   // BK+8: row stride 80B = 16B-aligned AND 20 banks (conflict-free)
#define LDB_S 66   // BN+2: row stride 132B = 33 banks (odd -> conflict-free)

__global__ __launch_bounds__(256) void gemm_bf16(
    const unsigned short* __restrict__ A, int lda,
    const unsigned short* __restrict__ Bw, int ldb,
    int K, int epi,
    const float* __restrict__ bias,
    float* __restrict__ Cf, int ldc,
    float* __restrict__ outp, int t, int last,
    float* __restrict__ hf, unsigned short* __restrict__ xh)
{
  __shared__ unsigned short As[2][BM * LDA_S];
  __shared__ unsigned short Bs[2][BK * LDB_S];   // Bs[k][n] (transposed stage)

  const int tid  = threadIdx.x;
  const int lane = tid & 31;
  const int wave = tid >> 5;
  const int n0 = blockIdx.x * BN;
  const int m0 = blockIdx.y * BM;
  const int wm = (wave & 3) * 16;     // wave's row offset in block
  const int wn = (wave >> 2) * 32;    // wave's col offset in block

  const int h  = lane >> 4;           // lane half
  const int ml = lane & 15;

  // staging assignments
  const int arow = tid >> 2;          // 0..63 (4 threads/row, 16B each)
  const int aseg = (tid & 3) * 8;     // k offset (shorts) within chunk
  const int brow = tid & 63;          // n 0..63
  const int bseg = (tid >> 6) * 8;    // k offset within chunk

  const unsigned short* agp = A + (long)(m0 + arow) * lda + aseg;  // + k0
  const unsigned short* bgp = Bw + (long)(n0 + brow) * ldb + bseg; // + k0

  v8f acc0 = {}; v8f acc1 = {};

  const int nChunks = K / BK;

  // ---- prologue: stage chunk 0 into buffer 0 ----
  {
    unsigned ldsA = (unsigned)(size_t)&As[0][arow * LDA_S + aseg];
    unsigned long long ga = (unsigned long long)(size_t)(agp);
    asm volatile("global_load_async_to_lds_b128 %0, %1, off"
                 :: "v"(ldsA), "v"(ga) : "memory");
    uint4 d = *(const uint4*)(bgp);
    asm volatile("s_wait_asynccnt 0x0" ::: "memory");
    const unsigned short* e = (const unsigned short*)&d;
    #pragma unroll
    for (int i = 0; i < 8; ++i)
      Bs[0][(bseg + i) * LDB_S + brow] = e[i];
  }
  __syncthreads();

  for (int c = 0; c < nChunks; ++c) {
    const int cur = c & 1, nxt = cur ^ 1;
    const bool more = (c + 1 < nChunks);
    uint4 breg;
    if (more) {
      const int kn = (c + 1) * BK;
      unsigned ldsA = (unsigned)(size_t)&As[nxt][arow * LDA_S + aseg];
      unsigned long long ga = (unsigned long long)(size_t)(agp + kn);
      asm volatile("global_load_async_to_lds_b128 %0, %1, off"
                   :: "v"(ldsA), "v"(ga) : "memory");
      breg = *(const uint4*)(bgp + kn);
    }

    // ---- A fragment (16x32 bf16): lane -> M, VGPR v -> K pair ----
    union { v16bf v; unsigned u[8]; } af;
    #pragma unroll
    for (int vv = 0; vv < 8; ++vv) {
      int kb = (vv < 4) ? (2 * vv + 8 * h) : (16 + 2 * (vv - 4) + 8 * h);
      af.u[vv] = *(const unsigned*)&As[cur][(wm + ml) * LDA_S + kb];
    }
    // ---- B fragments (32x16): lane -> N, VGPR j -> K pair ----
    #pragma unroll
    for (int nt = 0; nt < 2; ++nt) {
      union { v16bf v; unsigned u[8]; } bfr;
      #pragma unroll
      for (int j = 0; j < 8; ++j) {
        int kb = (j < 4) ? (2 * j + 8 * h) : (16 + 2 * (j - 4) + 8 * h);
        int col = wn + nt * 16 + ml;
        bfr.u[j] = (unsigned)Bs[cur][kb * LDB_S + col] | ((unsigned)Bs[cur][(kb + 1) * LDB_S + col] << 16);
      }
      if (nt == 0)
        acc0 = __builtin_amdgcn_wmma_f32_16x16x32_bf16(false, af.v, false, bfr.v,
                                                       (short)0, acc0, false, false);
      else
        acc1 = __builtin_amdgcn_wmma_f32_16x16x32_bf16(false, af.v, false, bfr.v,
                                                       (short)0, acc1, false, false);
    }

    if (more) {
      asm volatile("s_wait_asynccnt 0x0" ::: "memory");   // A(next) landed in LDS
      const unsigned short* e = (const unsigned short*)&breg;
      #pragma unroll
      for (int i = 0; i < 8; ++i)
        Bs[nxt][(bseg + i) * LDB_S + brow] = e[i];
      __syncthreads();
    }
  }

  // ---- epilogue (C/D layout: VGPR r -> M = r + 8*half, lane[3:0] -> N) ----
  #pragma unroll
  for (int nt = 0; nt < 2; ++nt) {
    v8f acc = nt ? acc1 : acc0;
    #pragma unroll
    for (int r = 0; r < 8; ++r) {
      int m = m0 + wm + r + 8 * h;
      int n = n0 + wn + nt * 16 + ml;
      float v = acc[r];
      if (epi == 0) {
        Cf[(long)m * ldc + n] = v + bias[n];
      } else if (epi == 1) {
        Cf[(long)m * ldc + n] = v;
      } else {
        float hv = tanhf(v);
        outp[((long)m * TT + t) * HDIM + n] = hv;      // output[b, t, :]
        xh[(long)m * KG + DIN + n] = f2bf(hv);         // recurrent input, next step
        if (last) hf[(long)m * HDIM + n] = hv;         // h_f
      }
    }
  }
}

// ---------------------------------------------------------------------------
// LSTM pointwise: gates[B,4H] + c -> cy, hy; hy(bf16) -> cat[:,1024:2048]
// ---------------------------------------------------------------------------
__global__ __launch_bounds__(256) void lstm_pointwise(
    const float* __restrict__ gates, float* __restrict__ c,
    unsigned short* __restrict__ cat, float* __restrict__ cf, int last)
{
  int idx = blockIdx.x * 256 + threadIdx.x;   // 0 .. B*H-1
  int b = idx >> 10, j = idx & 1023;
  const float* g = gates + (long)b * NG;
  float ig = g[j], fg = g[HDIM + j], gg = g[2 * HDIM + j], og = g[3 * HDIM + j];
  float cp = c[idx];
  float si = 1.f / (1.f + expf(-ig));
  float sf = 1.f / (1.f + expf(-fg));
  float so = 1.f / (1.f + expf(-og));
  float cy = sf * cp + si * tanhf(gg);
  float hy = so * tanhf(cy);
  c[idx] = cy;
  cat[(long)b * 2 * HDIM + HDIM + j] = f2bf(hy);
  if (last) cf[idx] = cy;
}

// ---------------------------------------------------------------------------
// Attention: scores -> masked softmax -> wctx; bf16 wctx -> cat[:,0:1024]
// One block per batch. ctx stays f32 (L2-resident, bandwidth-bound).
// ---------------------------------------------------------------------------
__global__ __launch_bounds__(256) void attn_kernel(
    const float* __restrict__ ctx, const unsigned char* __restrict__ mask,
    const float* __restrict__ q, unsigned short* __restrict__ cat)
{
  __shared__ float sc[TT];
  __shared__ float red[256];
  const int b = blockIdx.x;
  const int tid = threadIdx.x, lane = tid & 31, wave = tid >> 5;
  const float* cb = ctx + (long)b * TT * HDIM;
  const float* qb = q + (long)b * HDIM;

  // scores: one wave per row l, coalesced dot over H, shfl_xor reduce
  for (int l = wave; l < TT; l += 8) {
    const float* row = cb + (long)l * HDIM;
    float acc = 0.f;
    for (int k = lane; k < HDIM; k += 32) acc += row[k] * qb[k];
    #pragma unroll
    for (int off = 16; off > 0; off >>= 1) acc += __shfl_xor(acc, off, 32);
    if (lane == 0)
      sc[l] = mask[b * TT + l] ? -__builtin_inff() : acc;
  }
  __syncthreads();

  // block softmax over 256 scores
  float v = sc[tid];
  red[tid] = v; __syncthreads();
  for (int s = 128; s > 0; s >>= 1) {
    if (tid < s) red[tid] = fmaxf(red[tid], red[tid + s]);
    __syncthreads();
  }
  float mx = red[0]; __syncthreads();
  float e = expf(v - mx);
  red[tid] = e; __syncthreads();
  for (int s = 128; s > 0; s >>= 1) {
    if (tid < s) red[tid] += red[tid + s];
    __syncthreads();
  }
  float sum = red[0]; __syncthreads();
  sc[tid] = e / sum;
  __syncthreads();

  // wctx = probs @ ctx[b]  (coalesced across threads; 4 H-dims per thread)
  float a0 = 0.f, a1 = 0.f, a2 = 0.f, a3 = 0.f;
  for (int l = 0; l < TT; ++l) {
    float p = sc[l];
    const float* row = cb + (long)l * HDIM;
    a0 += p * row[tid];       a1 += p * row[tid + 256];
    a2 += p * row[tid + 512]; a3 += p * row[tid + 768];
  }
  unsigned short* cr = cat + (long)b * 2 * HDIM;
  cr[tid]       = f2bf(a0); cr[tid + 256] = f2bf(a1);
  cr[tid + 512] = f2bf(a2); cr[tid + 768] = f2bf(a3);
}

// ---------------------------------------------------------------------------
extern "C" void kernel_launch(void* const* d_in, const int* in_sizes, int n_in,
                              void* d_out, int out_size, void* d_ws, size_t ws_size,
                              hipStream_t stream) {
  const float* input = (const float*)d_in[0];
  const float* h0    = (const float*)d_in[1];
  const float* c0    = (const float*)d_in[2];
  const float* ctx   = (const float*)d_in[3];
  const unsigned char* mask = (const unsigned char*)d_in[4];
  const float* Wih   = (const float*)d_in[5];
  const float* Whh   = (const float*)d_in[6];
  const float* bih   = (const float*)d_in[7];
  const float* bhh   = (const float*)d_in[8];
  const float* Wattn = (const float*)d_in[9];
  const float* Wout  = (const float*)d_in[10];

  // ---- workspace carve-up (~23 MB) ----
  char* ws = (char*)d_ws;
  size_t off = 0;
  auto alloc = [&](size_t bytes) -> void* {
    void* p = ws + off;
    off = (off + bytes + 255) & ~(size_t)255;
    return p;
  };
  unsigned short* Wg   = (unsigned short*)alloc((size_t)NG * KG * 2);          // 12.6 MB
  unsigned short* Wa   = (unsigned short*)alloc((size_t)HDIM * HDIM * 2);      //  2 MB
  unsigned short* Wo   = (unsigned short*)alloc((size_t)HDIM * 2 * HDIM * 2);  //  4 MB
  float*          bias = (float*)alloc((size_t)NG * 4);
  unsigned short* xh   = (unsigned short*)alloc((size_t)BB * KG * 2);          // [x_t | h] bf16
  unsigned short* cat  = (unsigned short*)alloc((size_t)BB * 2 * HDIM * 2);    // [wctx | hy] bf16
  float*          gates = (float*)alloc((size_t)BB * NG * 4);
  float*          qbuf  = (float*)alloc((size_t)BB * HDIM * 4);
  float*          cstate = (float*)alloc((size_t)BB * HDIM * 4);

  float* out = (float*)d_out;
  float* hf = out + (size_t)BB * TT * HDIM;
  float* cf = hf + (size_t)BB * HDIM;

  pack_weights<<<2048, 256, 0, stream>>>(Wih, Whh, bih, bhh, Wattn, Wout,
                                         Wg, Wa, Wo, bias);
  init_state<<<(BB * HDIM) / 256, 256, 0, stream>>>(h0, c0, cstate, xh);

  for (int t = 0; t < TT; ++t) {
    int last = (t == TT - 1);
    prep_x<<<(BB * DIN) / 256, 256, 0, stream>>>(input, t, xh);
    // gates = [x_t|h] @ [W_ih|W_hh]^T + bias
    gemm_bf16<<<dim3(NG / BN, BB / BM), 256, 0, stream>>>(
        xh, KG, Wg, KG, KG, 0, bias, gates, NG,
        nullptr, 0, 0, nullptr, nullptr);
    lstm_pointwise<<<(BB * HDIM) / 256, 256, 0, stream>>>(gates, cstate, cat, cf, last);
    // q = hy @ W_attn^T   (A = hy slice of cat)
    gemm_bf16<<<dim3(HDIM / BN, BB / BM), 256, 0, stream>>>(
        cat + HDIM, 2 * HDIM, Wa, HDIM, HDIM, 1, nullptr, qbuf, HDIM,
        nullptr, 0, 0, nullptr, nullptr);
    attn_kernel<<<BB, 256, 0, stream>>>(ctx, mask, qbuf, cat);
    // h_tilde = tanh([wctx|hy] @ W_out^T) -> output, next-step h, h_f
    gemm_bf16<<<dim3(HDIM / BN, BB / BM), 256, 0, stream>>>(
        cat, 2 * HDIM, Wo, 2 * HDIM, 2 * HDIM, 2, nullptr, nullptr, 0,
        out, t, last, hf, xh);
  }
}
